// MHGRN_68736656605792
// MI455X (gfx1250) — compile-verified, matrix operand order checked by reading, not run on hardware
//
#include <hip/hip_runtime.h>
#include <hip/hip_bf16.h>
#include <stddef.h>

typedef __attribute__((ext_vector_type(16))) _Float16 v16h;
typedef __attribute__((ext_vector_type(8)))  _Float16 v8h;
typedef __attribute__((ext_vector_type(8)))  float    v8f;
typedef __attribute__((ext_vector_type(4)))  float    f32x4;

#define G_   64
#define NPG_ 320
#define R_   32
#define KHOP 3
#define H_   128
#define L_   1024
#define DEG_ 16
#define N_   (G_ * NPG_)    /* 20480 */
#define E_   (N_ * DEG_)    /* 327680 */

// Wave32 16-bit WMMA fragment layout (ISA 7.12.2): for lane l (hi = l>=16),
// half p maps to k = p + 8*hi + (p>=8 ? 8 : 0) within a 32-wide k-slab.
// => two contiguous 8-element runs per lane: [8*hi, 8*hi+8) and [16+8*hi, 24+8*hi).

// Build f16 A-fragment from f32 memory (two aligned 32B runs -> b128 loads).
__device__ __forceinline__ v16h make_a_frag(const float* __restrict__ slab, int hi) {
  const f32x4 c0 = *(const f32x4*)(slab + 8 * hi);
  const f32x4 c1 = *(const f32x4*)(slab + 8 * hi + 4);
  const f32x4 c2 = *(const f32x4*)(slab + 16 + 8 * hi);
  const f32x4 c3 = *(const f32x4*)(slab + 16 + 8 * hi + 4);
  v16h a;
  #pragma unroll
  for (int j = 0; j < 4; ++j) {
    a[j]      = (_Float16)c0[j];
    a[j + 4]  = (_Float16)c1[j];
    a[j + 8]  = (_Float16)c2[j];
    a[j + 12] = (_Float16)c3[j];
  }
  return a;
}

// Build f16 B-fragment from LDS-resident f16 row (two 16B runs -> ds_load_b128).
__device__ __forceinline__ v16h make_b_frag(const _Float16* sWrowSlab, int hi) {
  const v8h b0 = *(const v8h*)(sWrowSlab + 8 * hi);
  const v8h b1 = *(const v8h*)(sWrowSlab + 16 + 8 * hi);
  v16h b;
  #pragma unroll
  for (int j = 0; j < 8; ++j) { b[j] = b0[j]; b[j + 8] = b1[j]; }
  return b;
}

// ---------------------------------------------------------------------------
// Generic WMMA GEMM: Y[n, i] = sum_k Xrow(n)[k] * Wm[i, k]  (+ bias[i])
//   Xrow(n)[k] = (k < split) ? X[(n/rowdiv)*xstride + k] : X2[n*x2stride + k-split]
// split is a multiple of 32 so each 32-wide k-slab is entirely on one side.
// Block = 256 threads = 8 waves; wave owns a 16-node tile for one 16-col tile.
// ---------------------------------------------------------------------------
__global__ void __launch_bounds__(256) gemm_xwt_kernel(
    const float* __restrict__ X, int rowdiv, int xstride, int split,
    const float* __restrict__ X2, int x2stride,
    const float* __restrict__ Wm, const float* __restrict__ bias,
    float* __restrict__ Y, int ycols, int Kdim) {
  __shared__ _Float16 sW[16 * (L_ + H_)];   // up to Kdim = 1152 -> 36 KB
  const int i0 = blockIdx.y * 16;
  for (int idx = threadIdx.x; idx < 16 * Kdim; idx += 256) {
    int i = idx / Kdim, k = idx - i * Kdim;
    sW[i * Kdim + k] = (_Float16)Wm[(size_t)(i0 + i) * Kdim + k];
  }
  __syncthreads();

  const int wave = threadIdx.x >> 5;
  const int lane = threadIdx.x & 31;
  const int hi   = lane >> 4;
  const int lrow = lane & 15;
  const int nb   = blockIdx.x * 128 + wave * 16;
  const int node = nb + lrow;

  const float* xr  = X + (size_t)(node / rowdiv) * xstride;
  const float* x2r = X2 ? (X2 + (size_t)node * x2stride) : nullptr;
  const _Float16* sWrow = sW + lrow * Kdim;

  v8f acc;
  #pragma unroll
  for (int v = 0; v < 8; ++v) acc[v] = 0.0f;

  for (int k0 = 0; k0 < Kdim; k0 += 32) {
    const float* aslab = (k0 < split) ? (xr + k0) : (x2r + (k0 - split));
    const v16h a = make_a_frag(aslab, hi);
    const v16h b = make_b_frag(sWrow + k0, hi);
    acc = __builtin_amdgcn_wmma_f32_16x16x32_f16(false, a, false, b,
                                                 (short)0, acc, false, false);
  }

  const int col = i0 + lrow;
  const float bv = bias ? bias[col] : 0.0f;
  #pragma unroll
  for (int v = 0; v < 8; ++v) {
    const int m = v + (hi ? 8 : 0);
    Y[(size_t)(nb + m) * ycols + col] = acc[v] + bv;
  }
}

// ---------------------------------------------------------------------------
// Batched per-relation WMMA: M[n,r,:] = W_t[r] @ S[n,r,:]
// For fixed r:  M_r(N x H) = S_r(N x H) . W_t[r]^T.  blockIdx.y = r.
// W_t[r] staged in LDS as f16 (32 KB); each wave: 16-node tile x all 8 col
// tiles = 32 v_wmma, A-fragment loads amortized across the 8 column tiles.
// ---------------------------------------------------------------------------
__global__ void __launch_bounds__(256) rgemm_kernel(
    const float* __restrict__ S, const float* __restrict__ Wt,
    float* __restrict__ M) {
  __shared__ _Float16 sW[H_ * H_];          // 32 KB
  const int r = blockIdx.y;
  const float* Wr = Wt + (size_t)r * H_ * H_;
  for (int idx = threadIdx.x; idx < H_ * H_; idx += 256)
    sW[idx] = (_Float16)Wr[idx];
  __syncthreads();

  const int wave = threadIdx.x >> 5;
  const int lane = threadIdx.x & 31;
  const int hi   = lane >> 4;
  const int lrow = lane & 15;
  const int nb   = blockIdx.x * 128 + wave * 16;

  const float* srow = S + ((size_t)(nb + lrow) * R_ + r) * H_;

  v8f acc[8];
  #pragma unroll
  for (int it = 0; it < 8; ++it)
    #pragma unroll
    for (int v = 0; v < 8; ++v) acc[it][v] = 0.0f;

  for (int k0 = 0; k0 < H_; k0 += 32) {
    const v16h a = make_a_frag(srow + k0, hi);
    #pragma unroll
    for (int it = 0; it < 8; ++it) {
      const v16h b = make_b_frag(sW + (it * 16 + lrow) * H_ + k0, hi);
      acc[it] = __builtin_amdgcn_wmma_f32_16x16x32_f16(false, a, false, b,
                                                       (short)0, acc[it],
                                                       false, false);
    }
  }

  #pragma unroll
  for (int it = 0; it < 8; ++it)
    #pragma unroll
    for (int v = 0; v < 8; ++v) {
      const int m = v + (hi ? 8 : 0);
      M[((size_t)(nb + m) * R_ + r) * H_ + it * 16 + lrow] = acc[it][v];
    }
}

// ---------------------------------------------------------------------------
// Small helper kernels
// ---------------------------------------------------------------------------
__global__ void zero_kernel(float* __restrict__ p, size_t n) {
  for (size_t i = (size_t)blockIdx.x * blockDim.x + threadIdx.x; i < n;
       i += (size_t)gridDim.x * blockDim.x)
    p[i] = 0.0f;
}

__global__ void __launch_bounds__(256) expT_kernel(const float* __restrict__ rel_T,
                                                   float* __restrict__ expT) {
  int i = blockIdx.x * 256 + threadIdx.x;
  if (i < R_ * R_) expT[i] = expf(rel_T[i]);
}

__global__ void __launch_bounds__(128) beta_kernel(
    const int* __restrict__ node_type, const float* __restrict__ lm,
    const float* __restrict__ dst_w, const float* __restrict__ dst_b,
    float* __restrict__ beta) {
  __shared__ float red[128];
  const int n = blockIdx.x, tid = threadIdx.x;
  const int t = node_type[n];
  const float* s = lm + (size_t)(n / NPG_) * L_;
  const float* w = dst_w + (size_t)t * L_;
  float p = 0.0f;
  for (int k = tid; k < L_; k += 128) p += w[k] * s[k];
  red[tid] = p;
  __syncthreads();
  for (int st = 64; st > 0; st >>= 1) {
    if (tid < st) red[tid] += red[tid + st];
    __syncthreads();
  }
  if (tid == 0) beta[n] = red[0] + dst_b[t];
}

// hop-0 edge scatter: 32 threads/edge, 4 dims/thread (b128 gather + 4 atomics)
__global__ void __launch_bounds__(256) edgeA_kernel(
    const int* __restrict__ ei, const int* __restrict__ et,
    const float* __restrict__ gamma, const float* __restrict__ beta,
    const float* __restrict__ emb, float* __restrict__ S,
    float* __restrict__ D, float* __restrict__ geb) {
  const size_t idx = (size_t)blockIdx.x * 256 + threadIdx.x;
  const int e  = (int)(idx >> 5);
  const int h4 = (int)(idx & 31) * 4;
  const int s = ei[e], d = ei[E_ + e], r = et[e];
  const float g  = expf(gamma[(size_t)s * R_ + r]);
  const float w0 = g * expf(beta[d]);
  const size_t seg = (size_t)s * R_ + r;
  const f32x4 ev = *(const f32x4*)(emb + (size_t)d * H_ + h4);
  float* sp = S + seg * H_ + h4;
  atomicAdd(sp + 0, w0 * ev[0]);
  atomicAdd(sp + 1, w0 * ev[1]);
  atomicAdd(sp + 2, w0 * ev[2]);
  atomicAdd(sp + 3, w0 * ev[3]);
  if (h4 == 0) { atomicAdd(&D[seg], w0); geb[e] = g; }
}

// hop-1 edge scatter: S1[src*R+et] += ge * C[dst,et];  D1 += ge * Cd[dst,et]
__global__ void __launch_bounds__(256) edgeB_kernel(
    const int* __restrict__ ei, const int* __restrict__ et,
    const float* __restrict__ geb, const float* __restrict__ C,
    const float* __restrict__ Cd, float* __restrict__ S,
    float* __restrict__ D) {
  const size_t idx = (size_t)blockIdx.x * 256 + threadIdx.x;
  const int e  = (int)(idx >> 5);
  const int h4 = (int)(idx & 31) * 4;
  const int s = ei[e], d = ei[E_ + e], r = et[e];
  const float g = geb[e];
  const size_t sseg = (size_t)s * R_ + r;
  const size_t dseg = (size_t)d * R_ + r;
  const f32x4 cv = *(const f32x4*)(C + dseg * H_ + h4);
  float* sp = S + sseg * H_ + h4;
  atomicAdd(sp + 0, g * cv[0]);
  atomicAdd(sp + 1, g * cv[1]);
  atomicAdd(sp + 2, g * cv[2]);
  atomicAdd(sp + 3, g * cv[3]);
  if (h4 == 0) atomicAdd(&D[sseg], g * Cd[dseg]);
}

// C[n,q,h] = sum_r M[n,r,h] * expT[r,q]
__global__ void __launch_bounds__(256) cmix_kernel(
    const float* __restrict__ M, const float* __restrict__ expT,
    float* __restrict__ C) {
  __shared__ float sT[R_ * R_];
  for (int i = threadIdx.x; i < R_ * R_; i += 256) sT[i] = expT[i];
  __syncthreads();
  const size_t idx = (size_t)blockIdx.x * 256 + threadIdx.x;
  const int n = (int)(idx >> 7), h = (int)(idx & 127);
  float mv[R_];
  #pragma unroll
  for (int r = 0; r < R_; ++r) mv[r] = M[((size_t)n * R_ + r) * H_ + h];
  for (int q = 0; q < R_; ++q) {
    float acc = 0.0f;
    #pragma unroll
    for (int r = 0; r < R_; ++r) acc += mv[r] * sT[r * R_ + q];
    C[((size_t)n * R_ + q) * H_ + h] = acc;
  }
}

// Cd[n,q] = sum_r D[n,r] * expT[r,q]
__global__ void __launch_bounds__(256) cdmix_kernel(
    const float* __restrict__ D, const float* __restrict__ expT,
    float* __restrict__ Cd) {
  __shared__ float sT[R_ * R_];
  for (int i = threadIdx.x; i < R_ * R_; i += 256) sT[i] = expT[i];
  __syncthreads();
  const int idx = blockIdx.x * 256 + threadIdx.x;
  const int n = idx >> 5, q = idx & 31;
  float acc = 0.0f;
  #pragma unroll
  for (int r = 0; r < R_; ++r) acc += D[(size_t)n * R_ + r] * sT[r * R_ + q];
  Cd[idx] = acc;
}

// A[n,h] = sum_r t0[r]*M[n,r,h];  den[n] = sum_r t0[r]*D[n,r]  (t0 = expT[:,0])
__global__ void __launch_bounds__(256) aden_kernel(
    const float* __restrict__ M, const float* __restrict__ D,
    const float* __restrict__ expT, float* __restrict__ A,
    float* __restrict__ den) {
  __shared__ float t0[R_];
  if (threadIdx.x < R_) t0[threadIdx.x] = expT[threadIdx.x * R_];
  __syncthreads();
  const size_t idx = (size_t)blockIdx.x * 256 + threadIdx.x;
  const int n = (int)(idx >> 7), h = (int)(idx & 127);
  float acc = 0.0f;
  #pragma unroll
  for (int r = 0; r < R_; ++r) acc += t0[r] * M[((size_t)n * R_ + r) * H_ + h];
  A[idx] = acc;
  if (h == 0) {
    float dd = 0.0f;
    #pragma unroll
    for (int r = 0; r < R_; ++r) dd += t0[r] * D[(size_t)n * R_ + r];
    den[n] = dd;
  }
}

// in-place: num[n,h] = den[n] > 0 ? num/den : 0
__global__ void __launch_bounds__(256) zfin_kernel(float* __restrict__ num,
                                                   const float* __restrict__ den) {
  const size_t idx = (size_t)blockIdx.x * 256 + threadIdx.x;
  const float d = den[idx >> 7];
  const float v = num[idx];
  num[idx] = (d > 0.0f) ? (v / d) : 0.0f;
}

// P1 = W[2,0] @ W[1,0]   (128x128)
__global__ void __launch_bounds__(256) p1_kernel(const float* __restrict__ W2,
                                                 const float* __restrict__ W1,
                                                 float* __restrict__ P1) {
  const int idx = blockIdx.x * 256 + threadIdx.x;
  const int i = idx >> 7, j = idx & 127;
  float acc = 0.0f;
  for (int k = 0; k < H_; ++k) acc += W2[i * H_ + k] * W1[k * H_ + j];
  P1[idx] = acc;
}

// lmzAT[h,l] = lmz_A[l,h]
__global__ void __launch_bounds__(256) transpose_kernel(const float* __restrict__ A,
                                                        float* __restrict__ At) {
  const int idx = blockIdx.x * 256 + threadIdx.x;
  const int l = idx >> 7, h = idx & 127;
  At[(size_t)h * L_ + l] = A[idx];
}

// per-node 3-way softmax over path lengths; z1_0 == 0 and lmz_b cancels.
__global__ void __launch_bounds__(128) attn_kernel(
    const float* __restrict__ u, const float* __restrict__ z11,
    const float* __restrict__ z12, float* __restrict__ z2) {
  __shared__ float r1[128], r2[128], a_s[2];
  const int n = blockIdx.x, h = threadIdx.x;
  const size_t base = (size_t)n * H_ + h;
  const float uv = u[base], zv1 = z11[base], zv2 = z12[base];
  r1[h] = uv * zv1;
  r2[h] = uv * zv2;
  __syncthreads();
  for (int st = 64; st > 0; st >>= 1) {
    if (h < st) { r1[h] += r1[h + st]; r2[h] += r2[h + st]; }
    __syncthreads();
  }
  if (h == 0) {
    const float l1 = r1[0], l2 = r2[0];
    const float m  = fmaxf(0.0f, fmaxf(l1, l2));
    const float e0 = expf(-m), e1 = expf(l1 - m), e2 = expf(l2 - m);
    const float inv = 1.0f / (e0 + e1 + e2);
    a_s[0] = e1 * inv;
    a_s[1] = e2 * inv;
  }
  __syncthreads();
  z2[base] = a_s[0] * zv1 + a_s[1] * zv2;
}

// out[g,h] = max over the 320 nodes of group g of h2[n,h]
__global__ void __launch_bounds__(128) segmax_kernel(const float* __restrict__ h2,
                                                     float* __restrict__ out) {
  const int g = blockIdx.x, h = threadIdx.x;
  float m = -3.402823466e38f;
  for (int j = 0; j < NPG_; ++j)
    m = fmaxf(m, h2[((size_t)(g * NPG_ + j)) * H_ + h]);
  out[g * H_ + h] = m;
}

// ---------------------------------------------------------------------------
extern "C" void kernel_launch(void* const* d_in, const int* in_sizes, int n_in,
                              void* d_out, int out_size, void* d_ws, size_t ws_size,
                              hipStream_t stream) {
  (void)in_sizes; (void)n_in; (void)out_size; (void)ws_size;
  const int*   node_type  = (const int*)  d_in[1];
  const float* node_emb   = (const float*)d_in[2];
  const int*   edge_index = (const int*)  d_in[3];
  const int*   edge_type  = (const int*)  d_in[4];
  const float* lm_context = (const float*)d_in[5];
  const float* W          = (const float*)d_in[6];
  const float* rel_T      = (const float*)d_in[7];
  const float* rel_w      = (const float*)d_in[8];
  const float* rel_b      = (const float*)d_in[9];
  const float* dst_w      = (const float*)d_in[10];
  const float* dst_b      = (const float*)d_in[11];
  const float* lmz_A      = (const float*)d_in[12];
  const float* V_w        = (const float*)d_in[14];
  const float* V_b        = (const float*)d_in[15];
  float* out = (float*)d_out;

  // ---- workspace layout (floats). bufA: S0 -> C -> M1 ; bufB: M0 -> S1 ----
  float* ws = (float*)d_ws;
  size_t off = 0;
  const size_t NRH = (size_t)N_ * R_ * H_;
  float* bufA  = ws + off; off += NRH;
  float* bufB  = ws + off; off += NRH;
  float* gamma = ws + off; off += (size_t)N_ * R_;
  float* beta  = ws + off; off += N_;
  float* expT  = ws + off; off += R_ * R_;
  float* geb   = ws + off; off += E_;
  float* D0    = ws + off; off += (size_t)N_ * R_;
  float* D1    = ws + off; off += (size_t)N_ * R_;
  float* Cd    = ws + off; off += (size_t)N_ * R_;
  float* A1    = ws + off; off += (size_t)N_ * H_;
  float* A2    = ws + off; off += (size_t)N_ * H_;
  float* den1  = ws + off; off += N_;
  float* den2  = ws + off; off += N_;
  float* num1  = ws + off; off += (size_t)N_ * H_;   // becomes z1[1]
  float* num2  = ws + off; off += (size_t)N_ * H_;   // becomes z1[2]
  float* ubuf  = ws + off; off += (size_t)N_ * H_;
  float* z2    = ws + off; off += (size_t)N_ * H_;
  float* lmzAT = ws + off; off += (size_t)H_ * L_;
  float* P1    = ws + off; off += (size_t)H_ * H_;
  float* h2    = ws + off; off += (size_t)N_ * H_;

  const dim3 blk256(256), blk128(128);
  const int nTiles     = N_ / 128;              // 160
  const int edgeBlocks = (E_ * 32) / 256;       // 40960
  const int nhBlocks   = (N_ * H_) / 256;       // 10240

  // 1) zero accumulators (S0, D0)
  zero_kernel<<<2048, blk256, 0, stream>>>(bufA, NRH);
  zero_kernel<<<512,  blk256, 0, stream>>>(D0, (size_t)N_ * R_);

  // 2) gamma = s_node @ rel_w^T + rel_b  (WMMA, s_node virtual via rowdiv)
  gemm_xwt_kernel<<<dim3(nTiles, R_ / 16), blk256, 0, stream>>>(
      lm_context, NPG_, L_, L_, nullptr, 0, rel_w, rel_b, gamma, R_, L_);
  beta_kernel<<<N_, blk128, 0, stream>>>(node_type, lm_context, dst_w, dst_b, beta);
  expT_kernel<<<(R_ * R_ + 255) / 256, blk256, 0, stream>>>(rel_T, expT);

  // 3) hop-0 edge scatter -> S0 (bufA), D0
  edgeA_kernel<<<edgeBlocks, blk256, 0, stream>>>(
      edge_index, edge_type, gamma, beta, node_emb, bufA, D0, geb);

  // 4) M0 = per-r S0 @ W[0,r]^T  (WMMA) -> bufB
  rgemm_kernel<<<dim3(nTiles, R_), blk256, 0, stream>>>(bufA, W, bufB);

  // 5) A1/den1 from M0/D0 (needed before M0's buffer is recycled)
  aden_kernel<<<nhBlocks, blk256, 0, stream>>>(bufB, D0, expT, A1, den1);

  // 6) C = M0 . expT  -> bufA (overwrites S0);  Cd = D0 @ expT
  cmix_kernel<<<nhBlocks, blk256, 0, stream>>>(bufB, expT, bufA);
  cdmix_kernel<<<(N_ * R_) / 256, blk256, 0, stream>>>(D0, expT, Cd);

  // 7) zero S1 (bufB, M0 now dead) and D1; hop-1 edge scatter
  zero_kernel<<<2048, blk256, 0, stream>>>(bufB, NRH);
  zero_kernel<<<512,  blk256, 0, stream>>>(D1, (size_t)N_ * R_);
  edgeB_kernel<<<edgeBlocks, blk256, 0, stream>>>(
      edge_index, edge_type, geb, bufA, Cd, bufB, D1);

  // 8) M1 = per-r S1 @ W[1,r]^T  (WMMA) -> bufA (C now dead)
  rgemm_kernel<<<dim3(nTiles, R_), blk256, 0, stream>>>(
      bufB, W + (size_t)R_ * H_ * H_, bufA);
  aden_kernel<<<nhBlocks, blk256, 0, stream>>>(bufA, D1, expT, A2, den2);

  // 9) path projections: Ps[2] = W[2,0], Ps[1] = W[2,0] @ W[1,0]
  const float* W20 = W + (size_t)(2 * R_) * H_ * H_;
  const float* W10 = W + (size_t)(1 * R_) * H_ * H_;
  p1_kernel<<<(H_ * H_) / 256, blk256, 0, stream>>>(W20, W10, P1);

  // 10) num_k = A_k @ Ps[k]^T (WMMA), then z1_k = mask(num/den) in place
  gemm_xwt_kernel<<<dim3(nTiles, H_ / 16), blk256, 0, stream>>>(
      A1, 1, H_, H_, nullptr, 0, P1, nullptr, num1, H_, H_);
  gemm_xwt_kernel<<<dim3(nTiles, H_ / 16), blk256, 0, stream>>>(
      A2, 1, H_, H_, nullptr, 0, W20, nullptr, num2, H_, H_);
  zfin_kernel<<<nhBlocks, blk256, 0, stream>>>(num1, den1);
  zfin_kernel<<<nhBlocks, blk256, 0, stream>>>(num2, den2);

  // 11) u = s_node @ lmz_A  (WMMA with lmz_A^T)
  transpose_kernel<<<(L_ * H_) / 256, blk256, 0, stream>>>(lmz_A, lmzAT);
  gemm_xwt_kernel<<<dim3(nTiles, H_ / 16), blk256, 0, stream>>>(
      lm_context, NPG_, L_, L_, nullptr, 0, lmzAT, nullptr, ubuf, H_, L_);

  // 12) attention over path lengths -> z2
  attn_kernel<<<N_, blk128, 0, stream>>>(ubuf, num1, num2, z2);

  // 13) h2 = [s_node | z2] @ V_w^T + V_b  (WMMA, concat virtual: split=1024)
  gemm_xwt_kernel<<<dim3(nTiles, H_ / 16), blk256, 0, stream>>>(
      lm_context, NPG_, L_, L_, z2, H_, V_w, V_b, h2, H_, L_ + H_);

  // 14) per-group segment max -> output (64 x 128)
  segmax_kernel<<<G_, blk128, 0, stream>>>(h2, out);
}
